// TFM8_48971217109167
// MI455X (gfx1250) — matrix operand order repaired and checked by reading, not compiled, not run on hardware
//
#include <hip/hip_runtime.h>
#include <hip/hip_bf16.h>

// ---------------------------------------------------------------------------
// Problem constants (reference: B=4, C=256, H=W=64)
// ---------------------------------------------------------------------------
#define C_DIM 256
#define HW    4096          // 64*64
#define BATCH 4
#define NPIX  (BATCH * HW)  // 16384

// LDS staging geometry for k_attn (padded for bank-conflict-free ds_load_b128)
#define KROW 264            // 256 + 8 u16 pad  -> row stride 132 dwords
#define VROW 72             // 64  + 8 u16 pad  -> row stride 36 dwords
#define KBUF_U16 (64 * KROW)
#define VBUF_U16 (256 * VROW)
#define VBASE_U16 (2 * KBUF_U16)
#define SMEM_U16 (2 * KBUF_U16 + 2 * VBUF_U16)      // double-buffered K + V
#define SMEM_BYTES (SMEM_U16 * 2)                   // 141312 B (< 320 KB/WG)

// LDS staging geometry for k_qkv weight tiles (64 rows per phase, 4 phases)
#define WROW 264
#define WBUF_U16 (64 * WROW)
#define QKV_SMEM_BYTES (2 * WBUF_U16 * 2)           // 67584 B double-buffered

typedef __attribute__((ext_vector_type(16))) __bf16 v16bf;
typedef __attribute__((ext_vector_type(8)))  float  v8f;
typedef int v4i_vs __attribute__((__vector_size__(16)));   // matches builtin proto
typedef unsigned int       u32;
typedef unsigned short     u16;

// ---------------------------------------------------------------------------
// Helpers
// ---------------------------------------------------------------------------
__device__ __forceinline__ u16 f2bf(float f) {
    u32 u = __float_as_uint(f);
    u += 0x7FFFu + ((u >> 16) & 1u);     // round-to-nearest-even
    return (u16)(u >> 16);
}

__device__ __forceinline__ u32 pack_bf16(float a, float b) {
    u32 ua = __float_as_uint(a);
    u32 ub = __float_as_uint(b);
    ua += 0x7FFFu + ((ua >> 16) & 1u);
    ub += 0x7FFFu + ((ub >> 16) & 1u);
    return (ua >> 16) | (ub & 0xFFFF0000u);
}

// A operand (16x32 bf16, M=rows): lane m=lane&15, half=lane>>4.
// VGPR0-7 hold K = cb+16*half .. +15 (contiguous, matches B)
__device__ __forceinline__ v16bf load_tileA(const u16* __restrict__ row, int cb, int half) {
    union { v16bf v; uint4 q[2]; } u;
    u.q[0] = *(const uint4*)(row + cb + half * 16);
    u.q[1] = *(const uint4*)(row + cb + half * 16 + 8);
    return u.v;
}

// B operand (32x16 bf16): lane n=lane&15 fixed column; lanes 0-15 hold
// K = k0..k0+15, lanes 16-31 hold K = k0+16..k0+31 (pairs per VGPR).
__device__ __forceinline__ v16bf load_tileB(const u16* __restrict__ colrow, int k0, int half) {
    union { v16bf v; uint4 q[2]; } u;
    const uint4* p = (const uint4*)(colrow + k0 + half * 16);
    u.q[0] = p[0];
    u.q[1] = p[1];
    return u.v;
}

__device__ __forceinline__ v8f wmma_bf16(v16bf a, v16bf b, v8f c) {
    return __builtin_amdgcn_wmma_f32_16x16x32_bf16(false, a, false, b,
                                                   (short)0, c, false, false);
}

// ---- CDNA5 async global -> LDS copy (ASYNCcnt-tracked), with fallback ------
#if __has_builtin(__builtin_amdgcn_global_load_async_to_lds_b128)
#define HAVE_ASYNC_LDS 1
#else
#define HAVE_ASYNC_LDS 0
#endif

__device__ __forceinline__ void cp_async16(const u16* gsrc, u16* ldst) {
#if HAVE_ASYNC_LDS
    __builtin_amdgcn_global_load_async_to_lds_b128(
        (__attribute__((address_space(1))) v4i_vs*)(uintptr_t)gsrc,
        (__attribute__((address_space(3))) v4i_vs*)(u32)(uintptr_t)ldst,
        0, 0);
#else
    *(uint4*)ldst = *(const uint4*)gsrc;   // sync fallback (still DS-resident)
#endif
}

__device__ __forceinline__ void wait_async() {
#if __has_builtin(__builtin_amdgcn_s_wait_asynccnt)
    __builtin_amdgcn_s_wait_asynccnt(0);
#else
    asm volatile("s_wait_asynccnt 0x0" ::: "memory");
#endif
}

// Stage 64 rows x 256 u16 (32 KB) from a row-major bf16 matrix into LDS with
// padded rows. 2048 chunks of 16B, 8 per thread (256-thread block).
__device__ __forceinline__ void stage_rows64(const u16* __restrict__ src,
                                             u16* buf, int tid) {
#pragma unroll
    for (int i = 0; i < 8; ++i) {
        const int c   = tid + i * 256;
        const int row = c >> 4, seg = c & 15;
        cp_async16(src + row * C_DIM + seg * 8, buf + row * KROW + seg * 8);
    }
}

// ---------------------------------------------------------------------------
// Kernel 1: tiled transpose + f32->bf16:  x [B][C][HW] f32 -> xt [B*HW][C] bf16
// ---------------------------------------------------------------------------
__global__ void k_transpose_cvt(const float* __restrict__ x, u16* __restrict__ xt) {
    __shared__ float t[32][33];
    const int b   = blockIdx.z;
    const int hw0 = blockIdx.x * 32;
    const int c0  = blockIdx.y * 32;
    const int tx  = threadIdx.x;
    const int ty  = threadIdx.y;

    const float* xb = x + (size_t)b * C_DIM * HW;
#pragma unroll
    for (int r = 0; r < 32; r += 8)
        t[ty + r][tx] = xb[(size_t)(c0 + ty + r) * HW + hw0 + tx];
    __syncthreads();

    u16* xo = xt + (size_t)b * HW * C_DIM;
#pragma unroll
    for (int r = 0; r < 32; r += 8)
        xo[(size_t)(hw0 + ty + r) * C_DIM + c0 + tx] = f2bf(t[tx][ty + r]);
}

// ---------------------------------------------------------------------------
// Kernel 2: weight convert f32 -> bf16 (layout kept: [o][c])
// ---------------------------------------------------------------------------
__global__ void k_cvt_w(const float* __restrict__ w, u16* __restrict__ o, int n) {
    int i = blockIdx.x * blockDim.x + threadIdx.x;
    if (i < n) o[i] = f2bf(w[i]);
}

// ---------------------------------------------------------------------------
// Kernel 3: 1x1-conv GEMM via WMMA bf16.
// One wave = 16 pixels x 256 outputs; weights (identical for all 8 waves of
// the block) are staged in LDS in 4 async double-buffered phases of 64 rows,
// so B-operands come from ds_load_b128 instead of per-wave global loads.
// ---------------------------------------------------------------------------
__global__ void __launch_bounds__(256) k_qkv(const u16* __restrict__ Xt,
                                             const u16* __restrict__ Wb,
                                             const float* __restrict__ bias,
                                             u16* __restrict__ Out, int vmode) {
    extern __shared__ u16 wsm[];

    const int tid  = threadIdx.x;
    const int lane = tid & 31;
    const int wv   = tid >> 5;
    const int gw   = blockIdx.x * 8 + wv;
    const int p0   = gw * 16;
    const int half = lane >> 4;
    const int n    = lane & 15;

    v16bf A[8];
    {
        const u16* row = Xt + (size_t)(p0 + n) * C_DIM;
#pragma unroll
        for (int ks = 0; ks < 8; ++ks) A[ks] = load_tileA(row, ks * 32, half);
    }

    // prefetch weight rows 0..63 into buffer 0
    stage_rows64(Wb, wsm, tid);

#pragma unroll 1
    for (int p = 0; p < 4; ++p) {
        wait_async();
        __syncthreads();
        const int curb = p & 1;
        const int nxtb = curb ^ 1;
        if (p < 3)
            stage_rows64(Wb + (size_t)(p + 1) * 64 * C_DIM,
                         wsm + nxtb * WBUF_U16, tid);

        const u16* Wt = wsm + curb * WBUF_U16;
#pragma unroll
        for (int nt = 0; nt < 4; ++nt) {
            const int o0 = p * 64 + nt * 16;
            const u16* wrow = Wt + (nt * 16 + n) * WROW;
            v16bf Bt[8];
#pragma unroll
            for (int ks = 0; ks < 8; ++ks) Bt[ks] = load_tileB(wrow, ks * 32, half);

            v8f acc = {0.f, 0.f, 0.f, 0.f, 0.f, 0.f, 0.f, 0.f};
#pragma unroll
            for (int ks = 0; ks < 8; ++ks) acc = wmma_bf16(A[ks], Bt[ks], acc);

            const float bo = bias[o0 + n];
            if (!vmode) {
#pragma unroll
                for (int r = 0; r < 8; ++r) {
                    const int px = p0 + r + half * 8;
                    Out[(size_t)px * C_DIM + o0 + n] = f2bf(acc[r] + bo);
                }
            } else {
                const int b   = p0 / HW;
                const int hwl = (p0 - b * HW) + half * 8;
                u16* dst = Out + (size_t)b * C_DIM * HW + (size_t)(o0 + n) * HW + hwl;
                uint4 pk;
                pk.x = pack_bf16(acc[0] + bo, acc[1] + bo);
                pk.y = pack_bf16(acc[2] + bo, acc[3] + bo);
                pk.z = pack_bf16(acc[4] + bo, acc[5] + bo);
                pk.w = pack_bf16(acc[6] + bo, acc[7] + bo);
                *(uint4*)dst = pk;
            }
        }
        __syncthreads();   // all waves done with buffer p before it is reused
    }
}

// ---------------------------------------------------------------------------
// Kernel 4: fused attention + LayerNorm with async double-buffered LDS staging.
// Per key-group (64 keys): K tile 64x256 bf16 and V tile 256x64 bf16 are
// copied global->LDS once per workgroup (ASYNCcnt), shared by all 8 waves.
// ---------------------------------------------------------------------------
__device__ __forceinline__ void stage_group(const u16* __restrict__ Kbase,
                                            const u16* __restrict__ Vbase,
                                            u16* Kbuf, u16* Vbuf,
                                            int kl0, int tid) {
    // K: 64 rows x 256 u16 = 2048 chunks of 16B, 8 per thread
    stage_rows64(Kbase + (size_t)kl0 * C_DIM, Kbuf, tid);
    // V: 256 rows x 64 u16 = 2048 chunks of 16B, 8 per thread
    const u16* vg = Vbase + kl0;
#pragma unroll
    for (int i = 0; i < 8; ++i) {
        const int c   = tid + i * 256;
        const int row = c >> 3, seg = c & 7;
        cp_async16(vg + (size_t)row * HW + seg * 8, Vbuf + row * VROW + seg * 8);
    }
}

__global__ void __launch_bounds__(256, 1) k_attn(const u16* __restrict__ Qb,
                                                 const u16* __restrict__ Kb,
                                                 const u16* __restrict__ Vb,
                                                 const float* __restrict__ lnw,
                                                 const float* __restrict__ lnb,
                                                 float* __restrict__ out) {
    extern __shared__ u16 smem[];

    const int tid  = threadIdx.x;
    const int lane = tid & 31;
    const int wv   = tid >> 5;
    const int gw   = blockIdx.x * 8 + wv;
    const int q0   = gw * 16;
    const int b    = q0 / HW;
    const int qh   = q0 - b * HW;
    const int half = lane >> 4;
    const int n    = lane & 15;

    const u16* Kbase = Kb + (size_t)b * HW * C_DIM;   // [key][c]
    const u16* Vbase = Vb + (size_t)b * C_DIM * HW;   // [c][key]

    // Q resident as 8 B-operands (reused for all 4096 keys)
    v16bf Bq[8];
    {
        const u16* qrow = Qb + (size_t)(q0 + n) * C_DIM;
#pragma unroll
        for (int ks = 0; ks < 8; ++ks) Bq[ks] = load_tileB(qrow, ks * 32, half);
    }

    v8f O[16];
#pragma unroll
    for (int i = 0; i < 16; ++i) O[i] = (v8f){0.f, 0.f, 0.f, 0.f, 0.f, 0.f, 0.f, 0.f};

    // prefetch group 0 into buffer 0 (pointers computed, no pointer arrays)
    stage_group(Kbase, Vbase, smem, smem + VBASE_U16, 0, tid);

#pragma unroll 1
    for (int g = 0; g < 64; ++g) {
        wait_async();
        __syncthreads();                       // group g staged, all waves synced

        const int cur = g & 1;
        const int nxt = cur ^ 1;
        if (g < 63)                            // overlap copy of g+1 with compute
            stage_group(Kbase, Vbase,
                        smem + nxt * KBUF_U16,
                        smem + VBASE_U16 + nxt * VBUF_U16,
                        (g + 1) * 64, tid);

        const u16* Kbuf = smem + cur * KBUF_U16;
        const u16* Vbuf = smem + VBASE_U16 + cur * VBUF_U16;

        // ---- S^T = K x Q : 4 key-subtiles of 16, K-dim = 256 channels ----
        v8f St[4];
#pragma unroll
        for (int s = 0; s < 4; ++s) {
            const u16* krow = Kbuf + (s * 16 + n) * KROW;
            v16bf At[8];
#pragma unroll
            for (int ks = 0; ks < 8; ++ks) At[ks] = load_tileA(krow, ks * 32, half);
            v8f acc = {0.f, 0.f, 0.f, 0.f, 0.f, 0.f, 0.f, 0.f};
#pragma unroll
            for (int ks = 0; ks < 8; ++ks) acc = wmma_bf16(At[ks], Bq[ks], acc);
            St[s] = acc;
        }

        // ---- softmax over the 64 keys, per query column ----
        float mx = -3.4e38f;
#pragma unroll
        for (int s = 0; s < 4; ++s)
#pragma unroll
            for (int r = 0; r < 8; ++r) mx = fmaxf(mx, St[s][r]);
        mx = fmaxf(mx, __shfl_xor(mx, 16, 32));

        float sum = 0.f;
#pragma unroll
        for (int s = 0; s < 4; ++s)
#pragma unroll
            for (int r = 0; r < 8; ++r) {
                float e = __expf(St[s][r] - mx);
                St[s][r] = e;
                sum += e;
            }
        sum += __shfl_xor(sum, 16, 32);
        const float inv = 1.0f / sum;
#pragma unroll
        for (int s = 0; s < 4; ++s)
#pragma unroll
            for (int r = 0; r < 8; ++r) St[s][r] *= inv;

        // ---- convert P^T (C/D layout) into two 32x16 B operands ----
        v16bf Bp[2];
#pragma unroll
        for (int t = 0; t < 2; ++t) {
            float kv[16];
#pragma unroll
            for (int r = 0; r < 8; ++r) {
                float a0  = St[2 * t][r];
                float a1  = St[2 * t + 1][r];
                float a0x = __shfl_xor(a0, 16, 32);
                float a1x = __shfl_xor(a1, 16, 32);
                kv[r]     = half ? a1x : a0;
                kv[8 + r] = half ? a1  : a0x;
            }
            union { v16bf v; u32 u[8]; } ub;
#pragma unroll
            for (int j = 0; j < 8; ++j) ub.u[j] = pack_bf16(kv[2 * j], kv[2 * j + 1]);
            Bp[t] = ub.v;
        }

        // ---- O^T += V x P^T : 16 channel tiles, K-dim = 64 keys ----
#pragma unroll
        for (int ct = 0; ct < 16; ++ct) {
            const u16* vrow = Vbuf + (ct * 16 + n) * VROW;
            v16bf a0 = load_tileA(vrow, 0, half);
            v16bf a1 = load_tileA(vrow, 32, half);
            O[ct] = wmma_bf16(a0, Bp[0], O[ct]);
            O[ct] = wmma_bf16(a1, Bp[1], O[ct]);
        }
        __syncthreads();   // all waves done reading buf g before it is reused
    }

    // ---- fused LayerNorm over channels (held in registers) ----
    float s1 = 0.f, s2 = 0.f;
#pragma unroll
    for (int ct = 0; ct < 16; ++ct)
#pragma unroll
        for (int r = 0; r < 8; ++r) {
            float v = O[ct][r];
            s1 += v;
            s2 += v * v;
        }
    s1 += __shfl_xor(s1, 16, 32);
    s2 += __shfl_xor(s2, 16, 32);
    const float mean = s1 * (1.0f / 256.0f);
    const float var  = s2 * (1.0f / 256.0f) - mean * mean;
    const float rstd = rsqrtf(var + 1e-5f);

    const int hwq = qh + n;
    float* ob = out + (size_t)b * C_DIM * HW;
#pragma unroll
    for (int ct = 0; ct < 16; ++ct) {
        const int cb = ct * 16 + half * 8;
        float4 wa = *(const float4*)(lnw + cb);
        float4 wb = *(const float4*)(lnw + cb + 4);
        float4 ba = *(const float4*)(lnb + cb);
        float4 bb = *(const float4*)(lnb + cb + 4);
        float wsv[8] = {wa.x, wa.y, wa.z, wa.w, wb.x, wb.y, wb.z, wb.w};
        float bsv[8] = {ba.x, ba.y, ba.z, ba.w, bb.x, bb.y, bb.z, bb.w};
#pragma unroll
        for (int r = 0; r < 8; ++r) {
            const int c = cb + r;
            ob[(size_t)c * HW + hwq] = (O[ct][r] - mean) * rstd * wsv[r] + bsv[r];
        }
    }
}

// ---------------------------------------------------------------------------
// Launch
// ---------------------------------------------------------------------------
extern "C" void kernel_launch(void* const* d_in, const int* in_sizes, int n_in,
                              void* d_out, int out_size, void* d_ws, size_t ws_size,
                              hipStream_t stream) {
    const float* x1  = (const float*)d_in[0];
    const float* x2  = (const float*)d_in[1];
    const float* qw  = (const float*)d_in[2];
    const float* qb  = (const float*)d_in[3];
    const float* kw  = (const float*)d_in[4];
    const float* kb  = (const float*)d_in[5];
    const float* vw  = (const float*)d_in[6];
    const float* vb  = (const float*)d_in[7];
    const float* lnw = (const float*)d_in[8];
    const float* lnb = (const float*)d_in[9];
    float* out = (float*)d_out;

    // Workspace layout (bf16 = u16), total ~40.4 MB
    u16* X1t = (u16*)d_ws;                       // [NPIX][C]
    u16* X2t = X1t + (size_t)NPIX * C_DIM;
    u16* Qb  = X2t + (size_t)NPIX * C_DIM;       // [NPIX][C]
    u16* Kb  = Qb  + (size_t)NPIX * C_DIM;       // [NPIX][C]
    u16* Vb  = Kb  + (size_t)NPIX * C_DIM;       // [B][C][HW]
    u16* Wq  = Vb  + (size_t)NPIX * C_DIM;       // [C][C]
    u16* Wk  = Wq  + (size_t)C_DIM * C_DIM;
    u16* Wv  = Wk  + (size_t)C_DIM * C_DIM;

    dim3 tb(32, 8);
    dim3 tg(HW / 32, C_DIM / 32, BATCH);
    k_transpose_cvt<<<tg, tb, 0, stream>>>(x1, X1t);
    k_transpose_cvt<<<tg, tb, 0, stream>>>(x2, X2t);

    const int nW = C_DIM * C_DIM;
    k_cvt_w<<<(nW + 255) / 256, 256, 0, stream>>>(qw, Wq, nW);
    k_cvt_w<<<(nW + 255) / 256, 256, 0, stream>>>(kw, Wk, nW);
    k_cvt_w<<<(nW + 255) / 256, 256, 0, stream>>>(vw, Wv, nW);

    k_qkv<<<128, 256, QKV_SMEM_BYTES, stream>>>(X1t, Wq, qb, Qb, 0);
    k_qkv<<<128, 256, QKV_SMEM_BYTES, stream>>>(X2t, Wk, kb, Kb, 0);
    k_qkv<<<128, 256, QKV_SMEM_BYTES, stream>>>(X2t, Wv, vb, Vb, 1);

    k_attn<<<128, 256, SMEM_BYTES, stream>>>(Qb, Kb, Vb, lnw, lnb, out);

    (void)in_sizes; (void)n_in; (void)out_size; (void)ws_size;
}
